// MultiHeadSelfAttention_7241314861154
// MI455X (gfx1250) — matrix operand order repaired and checked by reading, not compiled, not run on hardware
//
#include <hip/hip_runtime.h>

// ---------------------------------------------------------------------------
// Multi-head self-attention forward for MI455X (gfx1250, wave32, WMMA bf16).
//   B=2, N=2048, E=1024, H=16, D=64
// Pipeline:
//   [0] convert x -> bf16, transpose-convert w_qkv / w_proj -> bf16
//   [1] qkv GEMM (32x64 wave tiles, all-bf16 fragments) + split/scale
//   [2] flash attention: 4 waves/block share one (b,h); K/V chunks staged
//       cooperatively in LDS (4x L2-traffic reduction), online softmax
//   [3] proj GEMM + bias
// All matrix math via v_wmma_f32_16x16x32_bf16 with f32 accumulation.
// ---------------------------------------------------------------------------

typedef __bf16 bf16_t;
typedef __attribute__((ext_vector_type(16))) __bf16 v16bf;
typedef __attribute__((ext_vector_type(8)))  float  v8f;

#define EMBED 1024
#define NHEAD 16
#define HDIM  64
#define SEQ   2048
#define BATCH 2
#define ROWS  (BATCH * SEQ)   // 4096

#define KS 72   // Ksh row stride (elements) - bank-spread padding
#define VS 40   // Vsh row stride (elements)

__device__ __forceinline__ bf16_t f2bf(float f) {
  unsigned u = __builtin_bit_cast(unsigned, f);
  u += 0x7FFFu + ((u >> 16) & 1u);            // round-to-nearest-even
  unsigned short h = (unsigned short)(u >> 16);
  return __builtin_bit_cast(bf16_t, h);
}

__device__ __forceinline__ v8f vzero8() {
  v8f z;
#pragma unroll
  for (int i = 0; i < 8; ++i) z[i] = 0.0f;
  return z;
}

__device__ __forceinline__ v8f wmma_bf16(v16bf a, v16bf b, v8f c) {
  return __builtin_amdgcn_wmma_f32_16x16x32_bf16(false, a, false, b,
                                                 (short)0, c, false, false);
}

// 16 contiguous bf16 (32B) -> v16bf. Used for B fragments (lane = N column,
// elements = 16 contiguous K values of this lane's half).
__device__ __forceinline__ v16bf load16(const bf16_t* p) {
  union { v16bf v; uint4 q[2]; } u;
  u.q[0] = *reinterpret_cast<const uint4*>(p);
  u.q[1] = *reinterpret_cast<const uint4*>(p + 8);
  return u.v;
}

// Copy 16 bf16 (32B) global -> LDS through registers.
__device__ __forceinline__ void copy16(bf16_t* dst, const bf16_t* src) {
  const uint4 u0 = *reinterpret_cast<const uint4*>(src);
  const uint4 u1 = *reinterpret_cast<const uint4*>(src + 8);
  reinterpret_cast<uint4*>(dst)[0] = u0;
  reinterpret_cast<uint4*>(dst)[1] = u1;
}

// A-matrix 16x32 bf16 fragment: lane = half*16 + m,
// elems 0..15 <-> K = half*16 + e (32B contiguous)
__device__ __forceinline__ v16bf load_a_bf16(const bf16_t* base, int stride,
                                             int lane) {
  const int m = lane & 15, half = lane >> 4;
  const bf16_t* p = base + (size_t)m * stride + half * 16;
  union { v16bf v; uint4 q[2]; } u;
  u.q[0] = *reinterpret_cast<const uint4*>(p);
  u.q[1] = *reinterpret_cast<const uint4*>(p + 8);
  return u.v;
}

// ---------------------------------------------------------------------------
// Kernel 0a: elementwise fp32 -> bf16 (for x). 4 elems / thread.
// ---------------------------------------------------------------------------
__global__ __launch_bounds__(256) void conv_kernel(const float* __restrict__ in,
                                                   bf16_t* __restrict__ out) {
  const size_t i = ((size_t)blockIdx.x * blockDim.x + threadIdx.x) * 4;
  const float4 v = *reinterpret_cast<const float4*>(in + i);
  out[i + 0] = f2bf(v.x);
  out[i + 1] = f2bf(v.y);
  out[i + 2] = f2bf(v.z);
  out[i + 3] = f2bf(v.w);
}

// ---------------------------------------------------------------------------
// Kernel 0b: transpose-convert weights: in fp32 [K][N] -> out bf16 [N][K].
// LDS-tiled 32x32 so both global sides stay coalesced.
// ---------------------------------------------------------------------------
__global__ __launch_bounds__(256) void tconv_kernel(const float* __restrict__ in,
                                                    bf16_t* __restrict__ out,
                                                    int K, int N) {
  __shared__ bf16_t tile[32][33];
  const int kb = blockIdx.x * 32, nb = blockIdx.y * 32;
  const int tx = threadIdx.x & 31, ty = threadIdx.x >> 5;  // ty: 0..7
#pragma unroll
  for (int i = 0; i < 32; i += 8)
    tile[ty + i][tx] = f2bf(in[(size_t)(kb + ty + i) * N + nb + tx]);
  __syncthreads();
#pragma unroll
  for (int i = 0; i < 32; i += 8)
    out[(size_t)(nb + ty + i) * K + kb + tx] = tile[tx][ty + i];
}

// ---------------------------------------------------------------------------
// Kernel 1: qkv = x @ w_qkv + b_qkv ; split into Q (scaled), K, V^T (bf16).
// Wave tile: 32(M) x 64(N). 128 mtiles * 48 ntiles = 6144 waves.
// ---------------------------------------------------------------------------
__global__ __launch_bounds__(256) void qkv_kernel(
    const bf16_t* __restrict__ Xb, const bf16_t* __restrict__ WT,
    const float* __restrict__ bias, bf16_t* __restrict__ Qb,
    bf16_t* __restrict__ Kb, bf16_t* __restrict__ Vt) {
  const int lane = threadIdx.x & 31;
  const int gw = (blockIdx.x * blockDim.x + threadIdx.x) >> 5;
  const int nt = gw % 48, mt = gw / 48;
  const int row0 = mt * 32, col0 = nt * 64;
  const int nl = lane & 15, half = lane >> 4;

  v8f acc[2][4];
#pragma unroll
  for (int r = 0; r < 2; ++r)
#pragma unroll
    for (int t = 0; t < 4; ++t) acc[r][t] = vzero8();

  const bf16_t* a0p = Xb + (size_t)row0 * EMBED;
  const bf16_t* a1p = Xb + (size_t)(row0 + 16) * EMBED;
  for (int k0 = 0; k0 < EMBED; k0 += 32) {
    if (k0 + 256 < EMBED) {  // stream prefetch (global_prefetch_b8)
      __builtin_prefetch(a0p + (size_t)nl * EMBED + k0 + 256, 0, 3);
      __builtin_prefetch(WT + (size_t)(col0 + lane) * EMBED + k0 + 256, 0, 3);
    }
    const v16bf a0 = load_a_bf16(a0p + k0, EMBED, lane);
    const v16bf a1 = load_a_bf16(a1p + k0, EMBED, lane);
#pragma unroll
    for (int t = 0; t < 4; ++t) {
      const v16bf b =
          load16(WT + (size_t)(col0 + t * 16 + nl) * EMBED + k0 + half * 16);
      acc[0][t] = wmma_bf16(a0, b, acc[0][t]);
      acc[1][t] = wmma_bf16(a1, b, acc[1][t]);
    }
  }

  // Epilogue. col0 is a multiple of 64 -> `which` (q/k/v) and head are
  // wave-uniform: hoist via readfirstlane so the split is a scalar branch.
  const int colu = __builtin_amdgcn_readfirstlane(col0);
  const int rowu = __builtin_amdgcn_readfirstlane(row0);
  const int which = colu >> 10;              // 0=q 1=k 2=v
  const int h = (colu & 1023) >> 6;          // head
#pragma unroll
  for (int r = 0; r < 2; ++r) {
#pragma unroll
    for (int t = 0; t < 4; ++t) {
      const int d0 = t * 16 + nl;            // d within head (0..63)
#pragma unroll
      for (int g = 0; g < 8; ++g) {
        const int row = rowu + r * 16 + half * 8 + g;
        const float val = acc[r][t][g] + bias[colu + t * 16 + nl];
        const int bb = row >> 11, n = row & (SEQ - 1);
        const size_t bh = (size_t)bb * NHEAD + h;
        if (which == 0)
          Qb[(bh * SEQ + n) * HDIM + d0] = f2bf(val * 0.125f);  // * D^-0.5
        else if (which == 1)
          Kb[(bh * SEQ + n) * HDIM + d0] = f2bf(val);
        else
          Vt[(bh * HDIM + d0) * SEQ + n] = f2bf(val);
      }
    }
  }
}

// ---------------------------------------------------------------------------
// Kernel 2: flash attention. 4 waves/block share one (b,h); each wave owns a
// 16-query tile (64 queries per block). Per 32-key chunk, K (32x64) and V^T
// (64x32) are staged cooperatively into LDS once and consumed by all 4 waves.
// ---------------------------------------------------------------------------
__global__ __launch_bounds__(128) void attn_kernel(
    const bf16_t* __restrict__ Qb, const bf16_t* __restrict__ Kb,
    const bf16_t* __restrict__ Vt, bf16_t* __restrict__ Ob) {
  __shared__ __align__(32) bf16_t Ksh[32][KS];      // keys x d   (padded)
  __shared__ __align__(32) bf16_t Vsh[64][VS];      // d    x key (padded)
  __shared__ __align__(32) bf16_t Pst[4][16 * 32];  // per-wave P tile

  const int tid = threadIdx.x;
  const int lane = tid & 31;
  const int wv = tid >> 5;
  const int bh = blockIdx.x >> 5;          // (b*16+h); 32 blocks per head
  const int q0 = (blockIdx.x & 31) * 64 + wv * 16;
  const int nl = lane & 15, half = lane >> 4;

  const bf16_t* Qh = Qb + ((size_t)bh * SEQ + q0) * HDIM;
  const bf16_t* Kh = Kb + (size_t)bh * SEQ * HDIM;
  const bf16_t* Vh = Vt + (size_t)bh * HDIM * SEQ;
  bf16_t* myP = &Pst[wv][0];

  const v16bf aq0 = load_a_bf16(Qh, HDIM, lane);       // d = 0..31
  const v16bf aq1 = load_a_bf16(Qh + 32, HDIM, lane);  // d = 32..63

  // cooperative-staging coordinates (128 threads)
  const int krow = tid >> 2, kseg = (tid & 3) * 16;  // K: 32 rows x 4 segs
  const int vrow = tid >> 1, vseg = (tid & 1) * 16;  // V: 64 rows x 2 segs

  float mrow[8], lrow[8];
#pragma unroll
  for (int g = 0; g < 8; ++g) { mrow[g] = -1e30f; lrow[g] = 0.0f; }
  v8f o[4];
#pragma unroll
  for (int t = 0; t < 4; ++t) o[t] = vzero8();

  for (int j0 = 0; j0 < SEQ; j0 += 32) {
    // ---- stage K chunk [j0..j0+32) x 64 and V^T chunk 64 x [j0..j0+32) ----
    __syncthreads();  // previous chunk's LDS reads are done
    copy16(&Ksh[krow][kseg], Kh + (size_t)(j0 + krow) * HDIM + kseg);
    copy16(&Vsh[vrow][vseg], Vh + (size_t)vrow * SEQ + j0 + vseg);
    __syncthreads();  // staged data visible to all waves

    // ---- S tiles for keys [j0, j0+16) and [j0+16, j0+32) ----
    v8f s0 = vzero8(), s1 = vzero8();
    s0 = wmma_bf16(aq0, load16(&Ksh[nl][half * 16]), s0);
    s0 = wmma_bf16(aq1, load16(&Ksh[nl][32 + half * 16]), s0);
    s1 = wmma_bf16(aq0, load16(&Ksh[16 + nl][half * 16]), s1);
    s1 = wmma_bf16(aq1, load16(&Ksh[16 + nl][32 + half * 16]), s1);

    // ---- online softmax over the 32-key chunk ----
#pragma unroll
    for (int g = 0; g < 8; ++g) {
      float cm = fmaxf(s0[g], s1[g]);
      cm = fmaxf(cm, __shfl_xor(cm, 1, 32));
      cm = fmaxf(cm, __shfl_xor(cm, 2, 32));
      cm = fmaxf(cm, __shfl_xor(cm, 4, 32));
      cm = fmaxf(cm, __shfl_xor(cm, 8, 32));   // row max within 16-lane group
      const float newm = fmaxf(mrow[g], cm);
      const float corr = __expf(mrow[g] - newm);
      mrow[g] = newm;
      const float p0 = __expf(s0[g] - newm);
      const float p1 = __expf(s1[g] - newm);
      float rs = p0 + p1;
      rs += __shfl_xor(rs, 1, 32);
      rs += __shfl_xor(rs, 2, 32);
      rs += __shfl_xor(rs, 4, 32);
      rs += __shfl_xor(rs, 8, 32);
      lrow[g] = lrow[g] * corr + rs;
#pragma unroll
      for (int t = 0; t < 4; ++t) o[t][g] *= corr;
      // stage P (row = half*8+g per C/D layout) as 16x32 row-major bf16
      myP[(half * 8 + g) * 32 + nl] = f2bf(p0);
      myP[(half * 8 + g) * 32 + 16 + nl] = f2bf(p1);
    }
    // P write -> read is same-wave: LDS ops are in-order per wave (DScnt);
    // just stop the compiler from reordering across this point.
    asm volatile("" ::: "memory");

    // ---- O += P[16x32] * V[32x64]  (4 WMMAs, B frags from LDS V^T) ----
    const v16bf pa = load_a_bf16(myP, 32, lane);
#pragma unroll
    for (int t = 0; t < 4; ++t)
      o[t] = wmma_bf16(pa, load16(&Vsh[t * 16 + nl][half * 16]), o[t]);
  }

  // ---- normalize and write attn-out as [B, N, E] bf16 ----
  const int bb = bh >> 4, h = bh & 15;
#pragma unroll
  for (int t = 0; t < 4; ++t) {
#pragma unroll
    for (int g = 0; g < 8; ++g) {
      const float val = o[t][g] / lrow[g];
      const int row = q0 + half * 8 + g;
      Ob[((size_t)bb * SEQ + row) * EMBED + h * HDIM + t * 16 + nl] = f2bf(val);
    }
  }
}

// ---------------------------------------------------------------------------
// Kernel 3: out = attn_out @ w_proj + b_proj. Wave tile 32x64.
// ---------------------------------------------------------------------------
__global__ __launch_bounds__(256) void proj_kernel(
    const bf16_t* __restrict__ A, const bf16_t* __restrict__ WT,
    const float* __restrict__ bias, float* __restrict__ out) {
  const int lane = threadIdx.x & 31;
  const int gw = (blockIdx.x * blockDim.x + threadIdx.x) >> 5;
  const int nt = gw & 15, mt = gw >> 4;   // 16 ntiles of 64, 128 mtiles of 32
  const int row0 = mt * 32, col0 = nt * 64;
  const int nl = lane & 15, half = lane >> 4;

  v8f acc[2][4];
#pragma unroll
  for (int r = 0; r < 2; ++r)
#pragma unroll
    for (int t = 0; t < 4; ++t) acc[r][t] = vzero8();

  const bf16_t* a0p = A + (size_t)row0 * EMBED;
  const bf16_t* a1p = A + (size_t)(row0 + 16) * EMBED;
  for (int k0 = 0; k0 < EMBED; k0 += 32) {
    if (k0 + 256 < EMBED) {
      __builtin_prefetch(a0p + (size_t)nl * EMBED + k0 + 256, 0, 3);
      __builtin_prefetch(WT + (size_t)(col0 + lane) * EMBED + k0 + 256, 0, 3);
    }
    const v16bf a0 = load_a_bf16(a0p + k0, EMBED, lane);
    const v16bf a1 = load_a_bf16(a1p + k0, EMBED, lane);
#pragma unroll
    for (int t = 0; t < 4; ++t) {
      const v16bf b =
          load16(WT + (size_t)(col0 + t * 16 + nl) * EMBED + k0 + half * 16);
      acc[0][t] = wmma_bf16(a0, b, acc[0][t]);
      acc[1][t] = wmma_bf16(a1, b, acc[1][t]);
    }
  }

#pragma unroll
  for (int r = 0; r < 2; ++r) {
#pragma unroll
    for (int t = 0; t < 4; ++t) {
#pragma unroll
      for (int g = 0; g < 8; ++g) {
        const int row = row0 + r * 16 + half * 8 + g;
        const int col = col0 + t * 16 + nl;
        out[(size_t)row * EMBED + col] = acc[r][t][g] + bias[col];
      }
    }
  }
}

// ---------------------------------------------------------------------------
extern "C" void kernel_launch(void* const* d_in, const int* in_sizes, int n_in,
                              void* d_out, int out_size, void* d_ws,
                              size_t ws_size, hipStream_t stream) {
  (void)in_sizes; (void)n_in; (void)out_size; (void)ws_size;
  const float* x      = (const float*)d_in[0];
  const float* w_qkv  = (const float*)d_in[1];
  const float* b_qkv  = (const float*)d_in[2];
  const float* w_proj = (const float*)d_in[3];
  const float* b_proj = (const float*)d_in[4];
  float* out = (float*)d_out;

  const size_t HE = (size_t)BATCH * NHEAD * SEQ * HDIM;  // 4M elements
  bf16_t* Xb    = (bf16_t*)d_ws;                        //  8 MB [4096][1024]
  bf16_t* WqkvT = Xb + (size_t)ROWS * EMBED;            //  6 MB [3072][1024]
  bf16_t* WprjT = WqkvT + (size_t)3 * EMBED * EMBED;    //  2 MB [1024][1024]
  bf16_t* Qb    = WprjT + (size_t)EMBED * EMBED;        //  8 MB
  bf16_t* Kb    = Qb + HE;                              //  8 MB
  bf16_t* Vt    = Kb + HE;                              //  8 MB
  bf16_t* AO    = Vt + HE;                              //  8 MB (48 MB total)

  // --- pre-pass: precision conversion (one-shot, ~2us of HBM traffic) ---
  conv_kernel<<<(ROWS * EMBED) / 1024, 256, 0, stream>>>(x, Xb);
  tconv_kernel<<<dim3(EMBED / 32, (3 * EMBED) / 32), 256, 0, stream>>>(
      w_qkv, WqkvT, EMBED, 3 * EMBED);
  tconv_kernel<<<dim3(EMBED / 32, EMBED / 32), 256, 0, stream>>>(
      w_proj, WprjT, EMBED, EMBED);

  // --- qkv GEMM: 6144 waves = 768 blocks x 8 waves ---
  qkv_kernel<<<768, 256, 0, stream>>>(Xb, WqkvT, b_qkv, Qb, Kb, Vt);
  // --- attention: 1024 blocks x 4 waves (one (b,h) + 64 queries / block) ---
  attn_kernel<<<1024, 128, 0, stream>>>(Qb, Kb, Vt, AO);
  // --- proj GEMM: 2048 waves = 256 blocks x 8 waves ---
  proj_kernel<<<256, 256, 0, stream>>>(AO, WprjT, b_proj, out);
}
